// Network_55499567399071
// MI455X (gfx1250) — compile-verified, matrix operand order checked by reading, not run on hardware
//
#include <hip/hip_runtime.h>
#include <hip/hip_bf16.h>
#include <math.h>

// ---------------------------------------------------------------------------
// Problem constants (from reference)
// ---------------------------------------------------------------------------
#define BB   32
#define SS   256
#define AA   6
#define HH   1024
#define OBS  520
#define GLOB 30
#define MASKN 52
#define AGF  73
#define NROW (BB*AA)          // 192 agent-batch rows
#define MTOT (NROW*SS)        // 49152 (GEMM rows, row = n*SS + s)
#define KVEC 102
#define KVECP 128             // padded to multiple of 32
#define DEMB 512
#define G3H  (3*HH)           // 3072
#define MBLK 4                // M-tiles per wave (64 rows) - B fragment reuse

typedef __bf16 bf16_t;
typedef __attribute__((ext_vector_type(16))) __bf16 v16bf;
typedef __attribute__((ext_vector_type(8)))  __bf16 v8bf;
typedef __attribute__((ext_vector_type(8)))  float  v8f;

// ---------------------------------------------------------------------------
// WMMA helpers: bf16 16x16x32, A row-major MxK, B given as row-major NxK (B^T)
// A-matrix lane layout (CDNA5 ISA 7.12.2, 16-bit A 16x32):
//   lanes 0-15  : row M=lane,    K = {0..7} then {16..23}
//   lanes 16-31 : row M=lane-16, K = {8..15} then {24..31}
// Both 8-element halves are contiguous in K -> two 16B loads per lane.
// B (32x16) mirrors this with N in place of M.
// C/D: v8f, element v -> M = v + (lane&16 ? 8 : 0), N = lane&15.
// ---------------------------------------------------------------------------
__device__ __forceinline__ v16bf load_rowK(const bf16_t* __restrict__ P,
                                           int row, int ld, int k0, int lane) {
  const int kb = (lane & 16) ? 8 : 0;
  const bf16_t* base = P + (size_t)row * ld + k0 + kb;
  v8bf lo = *(const v8bf*)(base);
  v8bf hi = *(const v8bf*)(base + 16);
  v16bf r;
#pragma unroll
  for (int i = 0; i < 8; ++i) { r[i] = lo[i]; r[8 + i] = hi[i]; }
  return r;
}

// One double-K-step of the 64x16 macro-tile: 20 VMEM reads (2xB + 8xA frags)
// then 8 WMMAs. sched_group_barrier pins "all loads first, then all WMMAs"
// so the fragments stay in distinct VGPRs and the waits become partial
// (descending loadcnt) instead of s_wait_loadcnt 0 before every WMMA.
// Essential for the low-occupancy GRU step GEMM (576 waves chip-wide).
__device__ __forceinline__ void k64_step(const bf16_t* __restrict__ A,
                                         const bf16_t* __restrict__ Bt,
                                         int rowA, int rowB, int K, int k0,
                                         int lane, v8f acc[MBLK]) {
  const v16bf b0 = load_rowK(Bt, rowB, K, k0, lane);
  const v16bf a00 = load_rowK(A, rowA,      K, k0, lane);
  const v16bf a01 = load_rowK(A, rowA + 16, K, k0, lane);
  const v16bf a02 = load_rowK(A, rowA + 32, K, k0, lane);
  const v16bf a03 = load_rowK(A, rowA + 48, K, k0, lane);
  const v16bf b1 = load_rowK(Bt, rowB, K, k0 + 32, lane);
  const v16bf a10 = load_rowK(A, rowA,      K, k0 + 32, lane);
  const v16bf a11 = load_rowK(A, rowA + 16, K, k0 + 32, lane);
  const v16bf a12 = load_rowK(A, rowA + 32, K, k0 + 32, lane);
  const v16bf a13 = load_rowK(A, rowA + 48, K, k0 + 32, lane);
  acc[0] = __builtin_amdgcn_wmma_f32_16x16x32_bf16(false, a00, false, b0, (short)0, acc[0], false, false);
  acc[1] = __builtin_amdgcn_wmma_f32_16x16x32_bf16(false, a01, false, b0, (short)0, acc[1], false, false);
  acc[2] = __builtin_amdgcn_wmma_f32_16x16x32_bf16(false, a02, false, b0, (short)0, acc[2], false, false);
  acc[3] = __builtin_amdgcn_wmma_f32_16x16x32_bf16(false, a03, false, b0, (short)0, acc[3], false, false);
  acc[0] = __builtin_amdgcn_wmma_f32_16x16x32_bf16(false, a10, false, b1, (short)0, acc[0], false, false);
  acc[1] = __builtin_amdgcn_wmma_f32_16x16x32_bf16(false, a11, false, b1, (short)0, acc[1], false, false);
  acc[2] = __builtin_amdgcn_wmma_f32_16x16x32_bf16(false, a12, false, b1, (short)0, acc[2], false, false);
  acc[3] = __builtin_amdgcn_wmma_f32_16x16x32_bf16(false, a13, false, b1, (short)0, acc[3], false, false);
#if __has_builtin(__builtin_amdgcn_sched_group_barrier)
  // group 1: 20 VMEM reads; group 2: 8 MFMA/WMMA (per K=64 chunk)
  __builtin_amdgcn_sched_group_barrier(0x0020, 20, 0);
  __builtin_amdgcn_sched_group_barrier(0x0008, 8, 0);
#endif
}

// ---------------------------------------------------------------------------
// Generic GEMM: C[M][N] = A[M][K](bf16) * Bt[N][K](bf16) + bias[N], optional
// ReLU, outputs f32 and/or bf16. One wave computes a 64x16 tile.
// Requires M % 64 == 0 and K % 64 == 0 (all call sites satisfy this).
// ---------------------------------------------------------------------------
__global__ void __launch_bounds__(128, 1)
gemm_nt_bf16(const bf16_t* __restrict__ A,
             const bf16_t* __restrict__ Bt,
             const float*  __restrict__ bias,
             float*  __restrict__ Cf,
             bf16_t* __restrict__ Cb,
             int M, int N, int K, int relu_flag) {
  const int wave  = blockIdx.x * (blockDim.x >> 5) + (threadIdx.x >> 5);
  const int ntil  = N >> 4;
  const int total = (M >> 6) * ntil;         // 64-row M blocks
  if (wave >= total) return;                 // uniform per-wave: EXEC stays full
  const int lane = threadIdx.x & 31;
  const int m0 = (wave / ntil) << 6;
  const int n0 = (wave % ntil) << 4;
  const int rowA = m0 + (lane & 15);
  const int rowB = n0 + (lane & 15);

  v8f acc[MBLK] = {{}, {}, {}, {}};
  for (int k0 = 0; k0 < K; k0 += 64)
    k64_step(A, Bt, rowA, rowB, K, k0, lane, acc);

  const int n = n0 + (lane & 15);
  const float bv = bias ? bias[n] : 0.f;
#pragma unroll
  for (int mi = 0; mi < MBLK; ++mi) {
    const int mb = m0 + 16 * mi + ((lane & 16) ? 8 : 0);
#pragma unroll
    for (int v = 0; v < 8; ++v) {
      float x = acc[mi][v] + bv;
      if (relu_flag) x = fmaxf(x, 0.f);
      size_t idx = (size_t)(mb + v) * N + n;
      if (Cf) Cf[idx] = x;
      if (Cb) Cb[idx] = (bf16_t)x;
    }
  }
}

// ---------------------------------------------------------------------------
// Embedding GEMM with per-row-group (type) weight/bias gather.
// A = vecb [MTOT][128], Bt = Wemb_b [5][512][128], C = emb_b [MTOT][512] bf16
// A 64-row M block stays within one agent row (256 % 64 == 0).
// ---------------------------------------------------------------------------
__global__ void __launch_bounds__(128, 1)
gemm_emb_bf16(const bf16_t* __restrict__ A,
              const bf16_t* __restrict__ Wemb,
              const float*  __restrict__ b_emb,
              const int*    __restrict__ types,
              bf16_t* __restrict__ Cb) {
  const int wave  = blockIdx.x * (blockDim.x >> 5) + (threadIdx.x >> 5);
  const int ntil  = DEMB >> 4;               // 32
  const int total = (MTOT >> 6) * ntil;
  if (wave >= total) return;
  const int lane = threadIdx.x & 31;
  const int m0 = (wave / ntil) << 6;
  const int n0 = (wave % ntil) << 4;
  const int t  = types[m0 / SS];
  const bf16_t* Bt = Wemb + (size_t)t * DEMB * KVECP;
  const int rowA = m0 + (lane & 15);
  const int rowB = n0 + (lane & 15);

  v8f acc[MBLK] = {{}, {}, {}, {}};
#pragma unroll
  for (int k0 = 0; k0 < KVECP; k0 += 64)
    k64_step(A, Bt, rowA, rowB, KVECP, k0, lane, acc);

  const int n = n0 + (lane & 15);
  const float bv = b_emb[t * DEMB + n];
#pragma unroll
  for (int mi = 0; mi < MBLK; ++mi) {
    const int mb = m0 + 16 * mi + ((lane & 16) ? 8 : 0);
#pragma unroll
    for (int v = 0; v < 8; ++v) {
      float x = fmaxf(acc[mi][v] + bv, 0.f);   // ReLU
      Cb[(size_t)(mb + v) * DEMB + n] = (bf16_t)x;
    }
  }
}

// ---------------------------------------------------------------------------
// Input parsing: vec (192,256,102) -> bf16, padded to K=128; extract types.
// vec[n][s][j] = x[b][s][a][j+1], x = [agent_a(73) | glob(30)].
// ---------------------------------------------------------------------------
__global__ void prep_vec(const float* __restrict__ inp,
                         bf16_t* __restrict__ vecb,
                         int* __restrict__ types) {
  const int row = blockIdx.x;                // 0..MTOT-1 : row = n*256+s
  const int j   = threadIdx.x;               // 0..127
  const int s = row % SS;
  const int n = row / SS;
  const int b = n / AA, a = n % AA;
  float v = 0.f;
  if (j < KVEC) {
    const int jj = j + 1;
    const size_t base = ((size_t)b * SS + s) * OBS;
    v = (jj < AGF) ? inp[base + GLOB + a * AGF + jj]
                   : inp[base + (jj - AGF)];
  }
  vecb[(size_t)row * KVECP + j] = (bf16_t)v;
  if (j == 0 && s == 0) types[n] = (int)rintf(v);
}

// f32 -> bf16 copy (same layout)
__global__ void cvt_bf16(const float* __restrict__ in, bf16_t* __restrict__ out,
                         size_t n) {
  for (size_t i = (size_t)blockIdx.x * blockDim.x + threadIdx.x; i < n;
       i += (size_t)gridDim.x * blockDim.x)
    out[i] = (bf16_t)in[i];
}

// in: [T][K][N] f32 -> out: [T][N][Kp] bf16 (zero-padded K)
__global__ void transpose_kn(const float* __restrict__ in,
                             bf16_t* __restrict__ out,
                             int T, int K, int N, int Kp) {
  const size_t total = (size_t)T * N * Kp;
  for (size_t i = (size_t)blockIdx.x * blockDim.x + threadIdx.x; i < total;
       i += (size_t)gridDim.x * blockDim.x) {
    const int k = (int)(i % Kp);
    const size_t r = i / Kp;
    const int nn = (int)(r % N);
    const int t  = (int)(r / N);
    out[i] = (bf16_t)((k < K) ? in[((size_t)t * K + k) * N + nn] : 0.f);
  }
}

__global__ void init_h(const float* __restrict__ s0, float* __restrict__ hf,
                       bf16_t* __restrict__ hb) {
  const int i = blockIdx.x * blockDim.x + threadIdx.x;
  if (i >= NROW * HH) return;
  const float v = s0[i];
  hf[i] = v; hb[i] = (bf16_t)v;
}

// ---------------------------------------------------------------------------
// GRU gate fusion for step t. xg bf16 [n*256+t][3072], hg f32 [192][3072].
// ---------------------------------------------------------------------------
__global__ void gru_gate(const bf16_t* __restrict__ xg,
                         const float*  __restrict__ hg,
                         float* __restrict__ hf, bf16_t* __restrict__ hb,
                         int t) {
  const int i = blockIdx.x * blockDim.x + threadIdx.x;
  if (i >= NROW * HH) return;
  const int n = i >> 10, j = i & (HH - 1);
  const size_t xr_ = ((size_t)n * SS + t) * G3H;
  const float xr = (float)xg[xr_ + j];
  const float xz = (float)xg[xr_ + HH + j];
  const float xn = (float)xg[xr_ + 2 * HH + j];
  const float* hgp = hg + (size_t)n * G3H;
  const float hr = hgp[j], hz = hgp[HH + j], hn = hgp[2 * HH + j];
  const float r = 1.f / (1.f + __expf(-(xr + hr)));
  const float z = 1.f / (1.f + __expf(-(xz + hz)));
  const float nn = tanhf(xn + r * hn);
  const float hv = (1.f - z) * nn + z * hf[i];
  hf[i] = hv; hb[i] = (bf16_t)hv;
}

// ---------------------------------------------------------------------------
// GAT attention: one block per (b, head); 6x6 attention over agents.
// hf layout: [192][1024], feature index = head*256 + d.
// ---------------------------------------------------------------------------
__global__ void gat_attn(const float* __restrict__ hf,
                         const float* __restrict__ a_s,
                         const float* __restrict__ a_d,
                         float* __restrict__ gf, bf16_t* __restrict__ gb) {
  const int b = blockIdx.x >> 2;
  const int h = blockIdx.x & 3;
  const int d = threadIdx.x;                 // 0..255
  __shared__ float ssrc[AA], sdst[AA], red[256];

  const float as = a_s[h * 256 + d];
  const float ad = a_d[h * 256 + d];
  float hv[AA];
#pragma unroll
  for (int i = 0; i < AA; ++i)
    hv[i] = hf[((size_t)(b * AA + i)) * HH + h * 256 + d];

  for (int i = 0; i < AA; ++i) {
    red[d] = hv[i] * as; __syncthreads();
    for (int o = 128; o > 0; o >>= 1) { if (d < o) red[d] += red[d + o]; __syncthreads(); }
    if (d == 0) ssrc[i] = red[0];
    __syncthreads();
    red[d] = hv[i] * ad; __syncthreads();
    for (int o = 128; o > 0; o >>= 1) { if (d < o) red[d] += red[d + o]; __syncthreads(); }
    if (d == 0) sdst[i] = red[0];
    __syncthreads();
  }

#pragma unroll
  for (int i = 0; i < AA; ++i) {
    float e[AA], mx = -1e30f;
#pragma unroll
    for (int j = 0; j < AA; ++j) {
      float v = ssrc[i] + sdst[j];
      e[j] = (v > 0.f) ? v : 0.2f * v;       // leaky_relu(0.2)
      mx = fmaxf(mx, e[j]);
    }
    float den = 0.f, o = 0.f;
#pragma unroll
    for (int j = 0; j < AA; ++j) { e[j] = __expf(e[j] - mx); den += e[j]; o += e[j] * hv[j]; }
    o /= den;
    const float out = (o > 0.f) ? o : (__expf(o) - 1.f);  // elu
    const size_t idx = ((size_t)(b * AA + i)) * HH + h * 256 + d;
    gf[idx] = out;
    if (gb) gb[idx] = (bf16_t)out;
  }
}

// ---------------------------------------------------------------------------
// Per-type MLP head: h1 = relu(f0 @ Wa1[pt] + ba1), logits = h1 @ Wa2[pt] + ba2
// ---------------------------------------------------------------------------
__global__ void head_kernel(const float* __restrict__ g1,
                            const int* __restrict__ types,
                            const float* __restrict__ Wa1,
                            const float* __restrict__ ba1,
                            const float* __restrict__ Wa2,
                            const float* __restrict__ ba2,
                            float* __restrict__ logits) {
  const int b = blockIdx.x;
  const int d = threadIdx.x;                 // 0..511
  __shared__ float sf0[HH];
  __shared__ float sh1[512];
  const int pt = types[b * AA];
  const float* f0 = g1 + (size_t)(b * AA) * HH;   // agent 0
  sf0[d] = f0[d]; sf0[512 + d] = f0[512 + d];
  __syncthreads();
  float acc = ba1[pt * 512 + d];
  const float* w1 = Wa1 + (size_t)pt * HH * 512;
  for (int f = 0; f < HH; ++f) acc += sf0[f] * w1[(size_t)f * 512 + d];
  sh1[d] = fmaxf(acc, 0.f);
  __syncthreads();
  if (d < MASKN) {
    float l = ba2[pt * MASKN + d];
    const float* w2 = Wa2 + (size_t)pt * 512 * MASKN;
    for (int f = 0; f < 512; ++f) l += sh1[f] * w2[f * MASKN + d];
    logits[b * MASKN + d] = l;
  }
}

__global__ void mask_kernel(const float* __restrict__ inp,
                            const float* __restrict__ logits,
                            float* __restrict__ out) {
  const int idx = blockIdx.x * blockDim.x + threadIdx.x;
  if (idx >= BB * SS * MASKN) return;
  const int c = idx % MASKN;
  const int bs = idx / MASKN;
  const int s = bs % SS, b = bs / SS;
  const float m = inp[((size_t)b * SS + s) * OBS + (OBS - MASKN) + c];
  out[idx] = (m != 0.f) ? logits[b * MASKN + c] : 0.f;
}

// ---------------------------------------------------------------------------
// Launch
// ---------------------------------------------------------------------------
static inline size_t align256(size_t x) { return (x + 255) & ~(size_t)255; }

extern "C" void kernel_launch(void* const* d_in, const int* in_sizes, int n_in,
                              void* d_out, int out_size, void* d_ws, size_t ws_size,
                              hipStream_t stream) {
  const float* inputs = (const float*)d_in[0];
  const float* state0 = (const float*)d_in[1];
  const float* W_emb  = (const float*)d_in[2];
  const float* b_emb  = (const float*)d_in[3];
  const float* W_ih   = (const float*)d_in[4];
  const float* W_hh   = (const float*)d_in[5];
  const float* b_ih   = (const float*)d_in[6];
  const float* b_hh   = (const float*)d_in[7];
  const float* Wg0    = (const float*)d_in[8];
  const float* a0s    = (const float*)d_in[9];
  const float* a0d    = (const float*)d_in[10];
  const float* Wg1    = (const float*)d_in[11];
  const float* a1s    = (const float*)d_in[12];
  const float* a1d    = (const float*)d_in[13];
  const float* Wa1    = (const float*)d_in[14];
  const float* ba1    = (const float*)d_in[15];
  const float* Wa2    = (const float*)d_in[16];
  const float* ba2    = (const float*)d_in[17];
  (void)in_sizes; (void)n_in; (void)out_size; (void)ws_size;

  // workspace carve-up
  char* ws = (char*)d_ws;
  size_t off = 0;
  auto take = [&](size_t bytes) { char* p = ws + off; off += align256(bytes); return p; };
  bf16_t* vecb   = (bf16_t*)take((size_t)MTOT * KVECP * 2);
  bf16_t* Wemb_b = (bf16_t*)take((size_t)5 * DEMB * KVECP * 2);
  bf16_t* Wih_b  = (bf16_t*)take((size_t)G3H * DEMB * 2);
  bf16_t* Whh_b  = (bf16_t*)take((size_t)G3H * HH * 2);
  bf16_t* Wg0_b  = (bf16_t*)take((size_t)HH * HH * 2);
  bf16_t* Wg1_b  = (bf16_t*)take((size_t)HH * HH * 2);
  bf16_t* emb_b  = (bf16_t*)take((size_t)MTOT * DEMB * 2);
  bf16_t* xg_b   = (bf16_t*)take((size_t)MTOT * G3H * 2);      // ~302 MB
  float*  hg_f   = (float*)take((size_t)NROW * G3H * 4);
  float*  h_f    = (float*)take((size_t)NROW * HH * 4);
  bf16_t* h_b    = (bf16_t*)take((size_t)NROW * HH * 2);
  int*    types  = (int*)take((size_t)NROW * 4);
  float*  hf_f   = (float*)take((size_t)NROW * HH * 4);
  float*  g0_f   = (float*)take((size_t)NROW * HH * 4);
  bf16_t* g0_b   = (bf16_t*)take((size_t)NROW * HH * 2);
  float*  g1_f   = (float*)take((size_t)NROW * HH * 4);
  float*  logits = (float*)take((size_t)BB * MASKN * 4);

  // ---- input prep + weight conversion ----
  prep_vec<<<MTOT, KVECP, 0, stream>>>(inputs, vecb, types);
  cvt_bf16<<<3072, 256, 0, stream>>>(W_ih, Wih_b, (size_t)G3H * DEMB);
  cvt_bf16<<<6144, 256, 0, stream>>>(W_hh, Whh_b, (size_t)G3H * HH);
  transpose_kn<<<1280, 256, 0, stream>>>(W_emb, Wemb_b, 5, KVEC, DEMB, KVECP);
  transpose_kn<<<4096, 256, 0, stream>>>(Wg0, Wg0_b, 1, HH, HH, HH);
  transpose_kn<<<4096, 256, 0, stream>>>(Wg1, Wg1_b, 1, HH, HH, HH);
  init_h<<<(NROW * HH + 255) / 256, 256, 0, stream>>>(state0, h_f, h_b);

  // ---- embedding: emb = relu(vec @ W_emb[type] + b_emb[type]) ----
  {
    int waves = (MTOT / 64) * (DEMB / 16);
    gemm_emb_bf16<<<(waves + 3) / 4, 128, 0, stream>>>(vecb, Wemb_b, b_emb, types, emb_b);
  }
  // ---- xg = emb @ W_ih^T + b_ih (one big WMMA GEMM, stored bf16) ----
  {
    int waves = (MTOT / 64) * (G3H / 16);
    gemm_nt_bf16<<<(waves + 3) / 4, 128, 0, stream>>>(emb_b, Wih_b, b_ih,
                                                      nullptr, xg_b,
                                                      MTOT, G3H, DEMB, 0);
  }
  // ---- GRU scan: per step, hg = h @ W_hh^T + b_hh then fused gates ----
  {
    const int waves = (NROW / 64) * (G3H / 16);        // 576
    const int gblk  = (waves + 3) / 4;
    const int eblk  = (NROW * HH + 255) / 256;
    for (int t = 0; t < SS; ++t) {
      gemm_nt_bf16<<<gblk, 128, 0, stream>>>(h_b, Whh_b, b_hh,
                                             hg_f, nullptr,
                                             NROW, G3H, HH, 0);
      gru_gate<<<eblk, 256, 0, stream>>>(xg_b, hg_f, h_f, h_b, t);
    }
  }
  // ---- hT output (second tuple element) ----
  hipMemcpyAsync((float*)d_out + (size_t)BB * SS * MASKN, h_f,
                 (size_t)NROW * HH * sizeof(float),
                 hipMemcpyDeviceToDevice, stream);

  // ---- GAT layer 0 ----
  {
    int waves = (NROW / 64) * (HH / 16);
    gemm_nt_bf16<<<(waves + 3) / 4, 128, 0, stream>>>(h_b, Wg0_b, nullptr,
                                                      hf_f, nullptr,
                                                      NROW, HH, HH, 0);
    gat_attn<<<BB * 4, 256, 0, stream>>>(hf_f, a0s, a0d, g0_f, g0_b);
  }
  // ---- GAT layer 1 ----
  {
    int waves = (NROW / 64) * (HH / 16);
    gemm_nt_bf16<<<(waves + 3) / 4, 128, 0, stream>>>(g0_b, Wg1_b, nullptr,
                                                      hf_f, nullptr,
                                                      NROW, HH, HH, 0);
    gat_attn<<<BB * 4, 256, 0, stream>>>(hf_f, a1s, a1d, g1_f, nullptr);
  }
  // ---- per-type head + mask broadcast ----
  head_kernel<<<BB, 512, 0, stream>>>(g1_f, types, Wa1, ba1, Wa2, ba2, logits);
  mask_kernel<<<(BB * SS * MASKN + 255) / 256, 256, 0, stream>>>(inputs, logits,
                                                                 (float*)d_out);
}
